// ScaledDotProductAttention_46231027974553
// MI455X (gfx1250) — compile-verified, matrix operand order checked by reading, not run on hardware
//
#include <hip/hip_runtime.h>
#include <hip/hip_bf16.h>

// ---------------------------------------------------------------------------
// ScaledDotProductAttention for MI455X (gfx1250, wave32, WMMA 16x16x32 f16)
//   B=4, S=4096, D=256
//   out = [context (B*S*D f32) | attn (B*S*S f32)]
//   ws  = qf16 (8MB) | kf16 (8MB) | vT f16 (8MB)
//
// Fused design: one workgroup (8 waves) owns a 16-query-row strip.
//   Phase 1: score strip (16 x 4096 f32) -> 256KB LDS  (WMMA f16, f32 acc)
//   Phase 2: row softmax in LDS (shfl reductions), NT-store attn to HBM
//   Phase 3: context = strip @ V from LDS (WMMA), no attn re-read from HBM
// ---------------------------------------------------------------------------

typedef __attribute__((ext_vector_type(16))) _Float16 v16h;
typedef __attribute__((ext_vector_type(8)))  _Float16 v8h;
typedef __attribute__((ext_vector_type(8)))  float    v8f;

#define ATTN_B 4
#define ATTN_S 4096
#define ATTN_D 256
#define STRIP_RS 4100   // padded LDS row stride in floats (4100 % 64 == 4)

// --- WMMA helper: D = A*B + C, f16 inputs, f32 accumulate ------------------
__device__ __forceinline__ v8f wmma_f16(v16h a, v16h b, v8f c) {
    return __builtin_amdgcn_wmma_f32_16x16x32_f16(
        /*neg_a=*/false, a, /*neg_b=*/false, b,
        /*c_mod=*/(short)0, c, /*reuse_a=*/false, /*reuse_b=*/false);
}

// Per-lane K pattern for 16x32 f16 A fragment (B column uses the same):
//   lanes 0-15 : K = {0..7, 16..23}, lanes 16-31 : K = {8..15, 24..31}
__device__ __forceinline__ v16h load_frag_h(const _Float16* __restrict__ row, int lane) {
    const int lo = (lane & 16) ? 8 : 0;
    v8h a = *reinterpret_cast<const v8h*>(row + lo);        // 16B load
    v8h b = *reinterpret_cast<const v8h*>(row + 16 + lo);   // 16B load
    v16h f;
#pragma unroll
    for (int i = 0; i < 8; ++i) { f[i] = a[i]; f[i + 8] = b[i]; }
    return f;
}

// Same fragment sourced from f32 (global or LDS), convert in-flight.
__device__ __forceinline__ v16h load_frag_f32(const float* row, int lane) {
    const int lo = (lane & 16) ? 8 : 0;
    v16h f;
#pragma unroll
    for (int i = 0; i < 8; ++i) {
        f[i]     = (_Float16)row[lo + i];
        f[i + 8] = (_Float16)row[16 + lo + i];
    }
    return f;
}

// ---------------------------------------------------------------------------
// 1) Projection: P[m,e] = sum_d X[m,d] * W[e,d] + bias[e], output f16.
//    NT GEMM, one 16x16 tile per wave, K=256 -> 8 WMMAs.
// ---------------------------------------------------------------------------
__global__ void proj_kernel(const float* __restrict__ X, const float* __restrict__ W,
                            const float* __restrict__ bias, _Float16* __restrict__ out) {
    const int M = ATTN_B * ATTN_S;              // 16384
    const int wave = (blockIdx.x * blockDim.x + threadIdx.x) >> 5;
    const int lane = threadIdx.x & 31;
    const int NT = ATTN_D / 16;                 // 16 N tiles
    const int tm = wave / NT;
    const int tn = wave % NT;
    if (tm >= M / 16) return;                   // wave-uniform

    const float* arow = X + (long)(tm * 16 + (lane & 15)) * ATTN_D;
    const float* brow = W + (long)(tn * 16 + (lane & 15)) * ATTN_D;

    v8f c = {};
#pragma unroll
    for (int k = 0; k < ATTN_D; k += 32) {
        v16h a = load_frag_f32(arow + k, lane);
        v16h b = load_frag_f32(brow + k, lane);
        c = wmma_f16(a, b, c);
    }
    const int n  = tn * 16 + (lane & 15);
    const float bv = bias[n];
    const int m0 = tm * 16 + ((lane >> 4) << 3);
#pragma unroll
    for (int v = 0; v < 8; ++v)
        out[(long)(m0 + v) * ATTN_D + n] = (_Float16)(c[v] + bv);
}

// ---------------------------------------------------------------------------
// 2) V transpose + f16 convert: VT[b][d][s] = (f16) V[b][s][d]  (LDS 32x32)
// ---------------------------------------------------------------------------
__global__ void vtrans_kernel(const float* __restrict__ V, _Float16* __restrict__ VT) {
    __shared__ _Float16 tile[32][33];
    const int b  = blockIdx.z;
    const int s0 = blockIdx.x * 32;
    const int d0 = blockIdx.y * 32;
    const int tx = threadIdx.x, ty = threadIdx.y;   // 32 x 8
#pragma unroll
    for (int i = ty; i < 32; i += 8)
        tile[i][tx] = (_Float16)V[((long)b * ATTN_S + s0 + i) * ATTN_D + d0 + tx];
    __syncthreads();
#pragma unroll
    for (int i = ty; i < 32; i += 8)
        VT[((long)b * ATTN_D + d0 + i) * ATTN_S + s0 + tx] = tile[tx][i];
}

// ---------------------------------------------------------------------------
// 3) Fused score + softmax + context per 16-row strip.
//    Block = 256 threads (8 waves). Dynamic LDS = 16 * STRIP_RS * 4 bytes.
// ---------------------------------------------------------------------------
__global__ void fused_attn_kernel(const _Float16* __restrict__ qf,
                                  const _Float16* __restrict__ kf,
                                  const _Float16* __restrict__ vT,
                                  const unsigned char* __restrict__ mask,
                                  float* __restrict__ attn,
                                  float* __restrict__ ctx) {
    extern __shared__ float strip[];            // [16][STRIP_RS]
    const int S = ATTN_S, D = ATTN_D;
    const int b    = blockIdx.x / (S / 16);
    const int tm   = blockIdx.x % (S / 16);
    const int t    = threadIdx.x;
    const int wave = t >> 5;
    const int lane = t & 31;
    const int l15  = lane & 15;
    const int mh   = (lane >> 4) << 3;          // M offset of this lane's C rows

    // ---- Phase 1: scores -> LDS strip. Wave w owns columns [w*512, w*512+512).
    const _Float16* arow = qf + ((long)b * S + tm * 16 + l15) * D;
    const float scale = 0.0625f;                // 1/sqrt(256)
#pragma unroll 1
    for (int tt = 0; tt < 8; ++tt) {
        const int col0 = wave * 512 + tt * 64;
        const _Float16* brow = kf + ((long)b * S + col0 + l15) * D;
        v8f c0 = {}, c1 = {}, c2 = {}, c3 = {};
#pragma unroll
        for (int k = 0; k < D; k += 32) {
            v16h a  = load_frag_h(arow + k, lane);
            v16h b0 = load_frag_h(brow + 0 * 16 * D + k, lane);
            v16h b1 = load_frag_h(brow + 1 * 16 * D + k, lane);
            v16h b2 = load_frag_h(brow + 2 * 16 * D + k, lane);
            v16h b3 = load_frag_h(brow + 3 * 16 * D + k, lane);
            c0 = wmma_f16(a, b0, c0);
            c1 = wmma_f16(a, b1, c1);
            c2 = wmma_f16(a, b2, c2);
            c3 = wmma_f16(a, b3, c3);
        }
#pragma unroll
        for (int v = 0; v < 8; ++v) {
            float* dst = strip + (mh + v) * STRIP_RS + col0 + l15;
            dst[0]  = c0[v] * scale;
            dst[16] = c1[v] * scale;
            dst[32] = c2[v] * scale;
            dst[48] = c3[v] * scale;
        }
    }
    __syncthreads();

    // ---- Phase 2: softmax. 16 threads per row; shfl-xor reductions (width 16).
    {
        const int r  = t >> 4;                  // row 0..15
        const int c0 = t & 15;                  // column phase
        float* row = strip + r * STRIP_RS;
        const bool on = mask[b * S + tm * 16 + r] != 0;

        float mx = -3.4e38f;
#pragma unroll 8
        for (int j = 0; j < 256; ++j) mx = fmaxf(mx, row[c0 + 16 * j]);
#pragma unroll
        for (int off = 1; off < 16; off <<= 1) mx = fmaxf(mx, __shfl_xor(mx, off, 16));

        float s = 0.0f;
#pragma unroll 8
        for (int j = 0; j < 256; ++j) {
            const float e = on ? __expf(row[c0 + 16 * j] - mx) : 1.0f;  // masked -> uniform
            row[c0 + 16 * j] = e;
            s += e;
        }
#pragma unroll
        for (int off = 1; off < 16; off <<= 1) s += __shfl_xor(s, off, 16);
        const float inv = 1.0f / s;

        float* orow = attn + (long)b * S * S + ((long)tm * 16 + r) * S;
#pragma unroll 8
        for (int j = 0; j < 256; ++j) {
            const float p = row[c0 + 16 * j] * inv;
            row[c0 + 16 * j] = p;                                   // keep for phase 3
            __builtin_nontemporal_store(p, orow + c0 + 16 * j);     // stream to HBM
        }
    }
    __syncthreads();

    // ---- Phase 3: context[16 x 256] = strip(16x4096) @ V. Wave w owns d-cols
    //      [w*32, w*32+32) as two 16-wide WMMA subtiles. A from LDS, B from L2.
    {
        const int d0 = wave * 32;
        const _Float16* brow0 = vT + ((long)b * D + d0 + l15) * S;
        const _Float16* brow1 = brow0 + 16L * S;
        v8f c0 = {}, c1 = {};
#pragma unroll 1
        for (int k = 0; k < S; k += 32) {
            v16h a  = load_frag_f32(strip + l15 * STRIP_RS + k, lane);  // ds_load + cvt
            v16h b0 = load_frag_h(brow0 + k, lane);
            v16h b1 = load_frag_h(brow1 + k, lane);
            c0 = wmma_f16(a, b0, c0);
            c1 = wmma_f16(a, b1, c1);
        }
#pragma unroll
        for (int v = 0; v < 8; ++v) {
            float* dst = ctx + ((long)b * S + tm * 16 + mh + v) * D + d0 + l15;
            dst[0]  = c0[v];
            dst[16] = c1[v];
        }
    }
}

// ---------------------------------------------------------------------------
extern "C" void kernel_launch(void* const* d_in, const int* in_sizes, int n_in,
                              void* d_out, int out_size, void* d_ws, size_t ws_size,
                              hipStream_t stream) {
    const float*         query = (const float*)d_in[0];
    const float*         key   = (const float*)d_in[1];
    const float*         value = (const float*)d_in[2];
    const unsigned char* mask  = (const unsigned char*)d_in[3];  // jax bool = 1 byte
    const float*         Wq    = (const float*)d_in[4];
    const float*         bq    = (const float*)d_in[5];
    const float*         Wk    = (const float*)d_in[6];
    const float*         bk    = (const float*)d_in[7];

    float* ctx  = (float*)d_out;                                  // B*S*D
    float* attn = ctx + (long)ATTN_B * ATTN_S * ATTN_D;           // B*S*S

    _Float16* qf = (_Float16*)d_ws;
    _Float16* kf = qf + (long)ATTN_B * ATTN_S * ATTN_D;
    _Float16* vT = kf + (long)ATTN_B * ATTN_S * ATTN_D;

    // 1) projections: 1024 M-tiles x 16 N-tiles = 16384 waves -> 2048 blocks
    proj_kernel<<<2048, 256, 0, stream>>>(query, Wq, bq, qf);
    proj_kernel<<<2048, 256, 0, stream>>>(key,   Wk, bk, kf);

    // 2) V transpose to f16 [D, S] per batch
    vtrans_kernel<<<dim3(ATTN_S / 32, ATTN_D / 32, ATTN_B), dim3(32, 8), 0, stream>>>(value, vT);

    // 3) fused score+softmax+context: one block per 16-row strip.
    //    Dynamic LDS: 16 rows * 4100 floats = 262,400 B (fits 320KB/WGP).
    const size_t lds_bytes = (size_t)16 * STRIP_RS * sizeof(float);
    fused_attn_kernel<<<ATTN_B * (ATTN_S / 16), 256, lds_bytes, stream>>>(
        qf, kf, vT, mask, attn, ctx);
}
